// GMambaBlock_7559142441632
// MI455X (gfx1250) — matrix operand order repaired and checked by reading, not compiled, not run on hardware
//
#include <hip/hip_runtime.h>
#include <math.h>

// ---------------------------------------------------------------------------
// GMambaBlock for MI455X (gfx1250, wave32, WMMA).
// GEMMs: f16 operands, f32 WMMA accumulation, 128x128 block tiles with
// double-buffered LDS staging via GLOBAL_LOAD_ASYNC_TO_LDS (ASYNCcnt).
// ---------------------------------------------------------------------------

typedef _Float16 half_t;
typedef __attribute__((ext_vector_type(16))) _Float16 v16h;
typedef __attribute__((ext_vector_type(8)))  _Float16 v8h;
typedef __attribute__((ext_vector_type(8)))  float    v8f;
typedef int v4i __attribute__((vector_size(16)));   // matches builtin param

#define MB_B   16
#define MB_L   1024
#define MB_D   512
#define MB_DI  1024
#define MB_DS  16
#define MB_DTR 32
#define MB_M   (MB_B * MB_L)   // 16384 flattened rows

// ---- CDNA5 async global->LDS path (guarded; falls back to ds_store path) ---
#if defined(__has_builtin)
#  if __has_builtin(__builtin_amdgcn_global_load_async_to_lds_b128)
#    define HAVE_ASYNC 1
#  endif
#endif
#ifndef HAVE_ASYNC
#  define HAVE_ASYNC 0
#endif

#if HAVE_ASYNC
__device__ __forceinline__ void async_cp16(const half_t* g, half_t* l) {
  // signature (from hipcc diagnostic): (int4 __device__*, int4 __shared__*, imm, imm)
  __builtin_amdgcn_global_load_async_to_lds_b128((v4i*)g, (v4i*)l, 0, 0);
}
__device__ __forceinline__ void wait_async0() {
#  if __has_builtin(__builtin_amdgcn_s_wait_asynccnt)
  __builtin_amdgcn_s_wait_asynccnt(0);
#  else
  asm volatile("s_wait_asynccnt 0x0" ::: "memory");
#  endif
}
#endif

// ---------------------------- activations ----------------------------------
__device__ __forceinline__ float act_apply(float v, int a) {
  if (a == 1) return v / (1.f + __expf(-v));             // silu
  if (a == 2) return 1.f / (1.f + __expf(-v));           // sigmoid
  if (a == 3) return (v > 20.f) ? v : log1pf(__expf(v)); // softplus
  return v;
}

// ------------------- WMMA fragment loaders (CDNA5 layouts) ------------------
// A 16x32 f16 (MxK): lane m=lane&15, g=lane>>4;
// halves 0..7 = K in [8g,8g+8), halves 8..15 = K in [16+8g,16+8g+8).
__device__ __forceinline__ v16h a_frag_from(const half_t* p /* row base + 8g */) {
  v8h lo = *(const v8h*)p;
  v8h hi = *(const v8h*)(p + 16);
  return __builtin_shufflevector(lo, hi, 0,1,2,3,4,5,6,7,8,9,10,11,12,13,14,15);
}

// ------------------------------ tiled WMMA GEMM -----------------------------
// C[M,N] = A[M,K] @ W[N,K]^T (+bias) (+res) (* (1+gate)); fused f32/f16 outs.
// Block = 8 waves covering BM x BN; K consumed in 32-slabs double-buffered in
// LDS, filled with async global->LDS copies.
template <int BM, int BN, int TM, int TN>
__global__ __launch_bounds__(256) void gemm_wmma_tiled(
    const half_t* __restrict__ A, int lda,
    const half_t* __restrict__ W, int ldb,
    const float* __restrict__ bias,
    const float* __restrict__ res,
    const float* __restrict__ gate,
    float* __restrict__ outf,
    half_t* __restrict__ o16a, int act16a,
    half_t* __restrict__ o16b, int act16b,
    int M, int N, int K) {
  constexpr int SK = 48;               // LDS row stride in halves (32 + pad)
  constexpr int WR = BM / (16 * TM);   // wave grid rows
  constexpr int WC = BN / (16 * TN);   // wave grid cols (WR*WC == 8)
  __shared__ __align__(64) half_t As[2 * BM * SK];
  __shared__ __align__(64) half_t Bs[2 * BN * SK];

  const int tid = threadIdx.x, lane = tid & 31, wave = tid >> 5;
  const int wr = wave % WR, wc = wave / WR;
  const int nTB = N / BN;
  const int bm0 = (blockIdx.x / nTB) * BM;
  const int bn0 = (blockIdx.x % nTB) * BN;

  // stage one 32-deep K slab of A (BM x 32) and W (BN x 32) into LDS buffer
  auto stage = [&](int buf, int k0) {
    half_t* Ad = As + buf * BM * SK;
    half_t* Bd = Bs + buf * BN * SK;
    for (int c = tid; c < BM * 4; c += 256) {            // 16B chunks
      int row = c >> 2, col = (c & 3) * 8;
      const half_t* gp = A + (size_t)(bm0 + row) * lda + k0 + col;
      half_t* lp = Ad + row * SK + col;
#if HAVE_ASYNC
      async_cp16(gp, lp);
#else
      *(uint4*)lp = *(const uint4*)gp;
#endif
    }
    for (int c = tid; c < BN * 4; c += 256) {
      int row = c >> 2, col = (c & 3) * 8;
      const half_t* gp = W + (size_t)(bn0 + row) * ldb + k0 + col;
      half_t* lp = Bd + row * SK + col;
#if HAVE_ASYNC
      async_cp16(gp, lp);
#else
      *(uint4*)lp = *(const uint4*)gp;
#endif
    }
  };

  v8f acc[TM][TN];
  for (int i = 0; i < TM; i++)
    for (int j = 0; j < TN; j++)
      acc[i][j] = (v8f){0.f, 0.f, 0.f, 0.f, 0.f, 0.f, 0.f, 0.f};

  const int nk = K / 32;
  stage(0, 0);
#if HAVE_ASYNC
  wait_async0();
#endif
  __syncthreads();

  const int m = lane & 15, g = lane >> 4;  // m doubles as n for B fragments
  for (int ki = 0; ki < nk; ki++) {
    const int cur = ki & 1;
    if (ki + 1 < nk) stage(cur ^ 1, (ki + 1) * 32);  // overlap next slab fill

    const half_t* Ab = As + cur * BM * SK;
    const half_t* Bb = Bs + cur * BN * SK;
    v16h af[TM], bf[TN];
    for (int i = 0; i < TM; i++)
      af[i] = a_frag_from(Ab + (wr * 16 * TM + i * 16 + m) * SK + 8 * g);
    for (int j = 0; j < TN; j++)
      bf[j] = a_frag_from(Bb + (wc * 16 * TN + j * 16 + m) * SK + 8 * g);
    for (int i = 0; i < TM; i++)
      for (int j = 0; j < TN; j++)
        acc[i][j] = __builtin_amdgcn_wmma_f32_16x16x32_f16(
            false, af[i], false, bf[j], (short)0, acc[i][j], false, false);

#if HAVE_ASYNC
    if (ki + 1 < nk) wait_async0();
#endif
    __syncthreads();
  }

  // fused epilogue
  for (int i = 0; i < TM; i++) {
    for (int j = 0; j < TN; j++) {
      const int tM = bm0 + wr * 16 * TM + i * 16;
      const int tN = bn0 + wc * 16 * TN + j * 16;
      const float bs = bias ? bias[tN + m] : 0.f;
      for (int r = 0; r < 8; r++) {
        const int mrow = tM + r + (g << 3);
        const size_t off = (size_t)mrow * N + tN + m;
        float v = acc[i][j][r] + bs;
        if (res)  v += res[off];
        if (gate) v *= (1.f + gate[off]);
        if (outf) outf[off] = v;
        if (o16a) o16a[off] = (half_t)act_apply(v, act16a);
        if (o16b) o16b[off] = (half_t)act_apply(v, act16b);
      }
    }
  }
}

// --------------------------- elementwise kernels ----------------------------
__global__ void k_cast16(const float* __restrict__ s, half_t* __restrict__ d, int n) {
  int i = blockIdx.x * blockDim.x + threadIdx.x;
  if (i < n) d[i] = (half_t)s[i];
}

__global__ void k_wc_split(const float* __restrict__ Wc, half_t* __restrict__ w0,
                           half_t* __restrict__ w1, half_t* __restrict__ w2, int n) {
  int i = blockIdx.x * blockDim.x + threadIdx.x;
  if (i < n) {
    w0[i] = (half_t)Wc[(size_t)i * 3 + 0];
    w1[i] = (half_t)Wc[(size_t)i * 3 + 1];
    w2[i] = (half_t)Wc[(size_t)i * 3 + 2];
  }
}

__global__ void k_prep(const float* __restrict__ x, half_t* __restrict__ xh,
                       float* __restrict__ flipf, half_t* __restrict__ fliph,
                       half_t* __restrict__ xm1, half_t* __restrict__ xp1, int n) {
  int i = blockIdx.x * blockDim.x + threadIdx.x;
  if (i >= n) return;
  int d = i & (MB_D - 1);
  int m = i >> 9;
  int l = m & (MB_L - 1);
  int b = m >> 10;
  xh[i] = (half_t)x[i];
  int ls = (l < 45) ? (44 - l) : l;
  float fv = x[((size_t)(b * MB_L + ls)) * MB_D + d];
  flipf[i] = fv;
  fliph[i] = (half_t)fv;
  xm1[i] = (l > 0)        ? (half_t)x[i - MB_D] : (half_t)0.f;
  xp1[i] = (l < MB_L - 1) ? (half_t)x[i + MB_D] : (half_t)0.f;
}

__global__ void k_act_pair(const float* __restrict__ s, half_t* __restrict__ si,
                           half_t* __restrict__ sg, int n) {
  int i = blockIdx.x * blockDim.x + threadIdx.x;
  if (i >= n) return;
  float v = s[i];
  float sgm = 1.f / (1.f + __expf(-v));
  if (si) si[i] = (half_t)(v * sgm);
  if (sg) sg[i] = (half_t)sgm;
}

__global__ void k_conv4(const half_t* __restrict__ xz, const float* __restrict__ cw,
                        const float* __restrict__ cb, half_t* __restrict__ xc, int n) {
  int i = blockIdx.x * blockDim.x + threadIdx.x;
  if (i >= n) return;
  int c = i & (MB_DI - 1);
  int m = i >> 10;
  int l = m & (MB_L - 1);
  int b = m >> 10;
  float acc = cb[c];
  for (int k = 0; k < 4; k++) {
    int ll = l - 3 + k;
    if (ll >= 0)
      acc += (float)xz[((size_t)(b * MB_L + ll)) * (2 * MB_DI) + c] * cw[c * 4 + k];
  }
  acc = acc / (1.f + __expf(-acc));
  xc[i] = (half_t)acc;
}

__global__ void k_combine(const float* __restrict__ cw, const float* __restrict__ mo,
                          const float* __restrict__ mf, const float* __restrict__ gr,
                          half_t* __restrict__ out, int n) {
  int i = blockIdx.x * blockDim.x + threadIdx.x;
  if (i < n) out[i] = (half_t)(cw[2] * mo[i] + cw[1] * mf[i] + cw[0] * gr[i]);
}

// ------------------------- Mamba selective scan -----------------------------
__global__ __launch_bounds__(1024) void k_mamba_scan(
    const float* __restrict__ projf,   // [M,64]: cols 32:48 = B, 48:64 = C
    const half_t* __restrict__ dt16,   // [M,DI] softplus(dt)
    const half_t* __restrict__ xch,    // [M,DI] silu(conv)
    const half_t* __restrict__ xz16,   // [M,2*DI] (z = cols DI:2DI)
    const float* __restrict__ A_log,   // [DI,DS]
    const float* __restrict__ Dp,      // [DI]
    half_t* __restrict__ y16) {        // [M,DI]
  const int b = blockIdx.x;
  const int d = threadIdx.x;
  __shared__ float bc_sh[2 * MB_DS];
  float An[MB_DS];
  for (int s = 0; s < MB_DS; s++) An[s] = -__expf(A_log[d * MB_DS + s]);
  const float Dv = Dp[d];
  float h[MB_DS];
  for (int s = 0; s < MB_DS; s++) h[s] = 0.f;

  for (int l = 0; l < MB_L; l++) {
    const size_t row = (size_t)b * MB_L + l;
    if (d < 2 * MB_DS) bc_sh[d] = projf[row * 64 + MB_DTR + d];
    __syncthreads();
    const float dt  = (float)dt16[row * MB_DI + d];
    const float xv  = (float)xch[row * MB_DI + d];
    const float dtx = dt * xv;
    float y = 0.f;
    for (int s = 0; s < MB_DS; s++) {
      const float dA = __expf(dt * An[s]);
      h[s] = dA * h[s] + dtx * bc_sh[s];
      y += h[s] * bc_sh[MB_DS + s];
    }
    const float z  = (float)xz16[row * (2 * MB_DI) + MB_DI + d];
    const float sz = z / (1.f + __expf(-z));
    y16[row * MB_DI + d] = (half_t)((y + xv * Dv) * sz);
    __syncthreads();
  }
}

// ------------------------------- GRU scan -----------------------------------
// Persistent single workgroup (8 waves). h (16x512 f16) and gh (16x1536 f16)
// live in LDS; per timestep gh = h @ Whh^T via WMMA with A read from LDS.
__global__ __launch_bounds__(256) void k_gru(
    const half_t* __restrict__ Whh,    // [1536,512] f16
    const half_t* __restrict__ gi16,   // [M,1536] f16
    float* __restrict__ gruout) {      // [M,512] f32
  __shared__ __align__(64) half_t hbuf[MB_B * MB_D];
  __shared__ __align__(64) half_t ghbuf[MB_B * 3 * MB_D];
  const int tid = threadIdx.x, lane = tid & 31, wave = tid >> 5;

  for (int i = tid; i < MB_B * MB_D; i += 256) hbuf[i] = (half_t)0.f;
  __syncthreads();

  for (int l = 0; l < MB_L; l++) {
    for (int j = 0; j < 12; j++) {           // 96 tiles of gh, 12 per wave
      const int n0 = (wave + 8 * j) << 4;
      v8f c = {0.f, 0.f, 0.f, 0.f, 0.f, 0.f, 0.f, 0.f};
      for (int k0 = 0; k0 < MB_D; k0 += 32) {
        const int m = lane & 15, g = lane >> 4;
        v16h a = a_frag_from(&hbuf[m * MB_D + k0 + 8 * g]);
        v16h bfr = a_frag_from(Whh + (size_t)(n0 + m) * MB_D + k0 + 8 * g);
        c = __builtin_amdgcn_wmma_f32_16x16x32_f16(false, a, false, bfr,
                                                   (short)0, c, false, false);
      }
      const int n = lane & 15, g = lane >> 4;
      for (int r = 0; r < 8; r++)
        ghbuf[(r + (g << 3)) * (3 * MB_D) + n0 + n] = (half_t)c[r];
    }
    __syncthreads();
    for (int i = tid; i < MB_B * MB_D; i += 256) {
      const int d = i & (MB_D - 1);
      const int bb = i >> 9;
      const size_t gir = ((size_t)(bb * MB_L + l)) * (3 * MB_D);
      const int ghr = bb * (3 * MB_D);
      const float rv = 1.f / (1.f + __expf(-((float)gi16[gir + d] +
                                             (float)ghbuf[ghr + d])));
      const float zv = 1.f / (1.f + __expf(-((float)gi16[gir + MB_D + d] +
                                             (float)ghbuf[ghr + MB_D + d])));
      const float nv = tanhf((float)gi16[gir + 2 * MB_D + d] +
                             rv * (float)ghbuf[ghr + 2 * MB_D + d]);
      const float hold = (float)hbuf[i];
      const float hn = (1.f - zv) * nv + zv * hold;
      hbuf[i] = (half_t)hn;
      gruout[((size_t)(bb * MB_L + l)) * MB_D + d] = hn;
    }
    __syncthreads();
  }
}

// ------------------------------- host side ----------------------------------
static void gemm(hipStream_t s, const half_t* A, int lda, const half_t* W, int ldb,
                 const float* bias, const float* res, const float* gate,
                 float* outf, half_t* o16a, int a16a, half_t* o16b, int a16b,
                 int M, int N, int K) {
  if ((N % 128) == 0) {
    int blocks = (M / 128) * (N / 128);
    gemm_wmma_tiled<128, 128, 2, 4><<<blocks, 256, 0, s>>>(
        A, lda, W, ldb, bias, res, gate, outf, o16a, a16a, o16b, a16b, M, N, K);
  } else {  // N multiple of 64 (x_proj)
    int blocks = (M / 64) * (N / 64);
    gemm_wmma_tiled<64, 64, 1, 2><<<blocks, 256, 0, s>>>(
        A, lda, W, ldb, bias, res, gate, outf, o16a, a16a, o16b, a16b, M, N, K);
  }
}

extern "C" void kernel_launch(void* const* d_in, const int* in_sizes, int n_in,
                              void* d_out, int out_size, void* d_ws, size_t ws_size,
                              hipStream_t stream) {
  const float* x       = (const float*)d_in[0];
  const float* cw      = (const float*)d_in[1];
  const float* W1      = (const float*)d_in[2];
  const float* b1      = (const float*)d_in[3];
  const float* W2      = (const float*)d_in[4];
  const float* b2      = (const float*)d_in[5];
  const float* Wp      = (const float*)d_in[6];
  const float* bp      = (const float*)d_in[7];
  const float* Wc      = (const float*)d_in[8];
  const float* bc      = (const float*)d_in[9];
  const float* Wsig    = (const float*)d_in[10];
  const float* bsig    = (const float*)d_in[11];
  const float* Wsi     = (const float*)d_in[12];
  const float* bsi     = (const float*)d_in[13];
  const float* Wih     = (const float*)d_in[14];
  const float* Whh     = (const float*)d_in[15];
  const float* inprojW = (const float*)d_in[16];
  const float* convW   = (const float*)d_in[17];
  const float* convb   = (const float*)d_in[18];
  const float* xprojW  = (const float*)d_in[19];
  const float* dtW     = (const float*)d_in[20];
  const float* dtb     = (const float*)d_in[21];
  const float* Alog    = (const float*)d_in[22];
  const float* Dp      = (const float*)d_in[23];
  const float* outprojW= (const float*)d_in[24];
  float* out = (float*)d_out;

  const int M = MB_M;
  char* base = (char*)d_ws;
  size_t off = 0;
  auto alloc = [&](size_t bytes) -> void* {
    void* p = base + off;
    off += (bytes + 255) & ~(size_t)255;
    return p;
  };
  // f16 weights
  half_t* w1h   = (half_t*)alloc(512 * 512 * 2);
  half_t* w2h   = (half_t*)alloc(512 * 512 * 2);
  half_t* wph   = (half_t*)alloc(512 * 512 * 2);
  half_t* wsigh = (half_t*)alloc(512 * 512 * 2);
  half_t* wsih  = (half_t*)alloc(512 * 512 * 2);
  half_t* wc0h  = (half_t*)alloc(512 * 512 * 2);
  half_t* wc1h  = (half_t*)alloc(512 * 512 * 2);
  half_t* wc2h  = (half_t*)alloc(512 * 512 * 2);
  half_t* wihh  = (half_t*)alloc(1536 * 512 * 2);
  half_t* whhh  = (half_t*)alloc(1536 * 512 * 2);
  half_t* inph  = (half_t*)alloc(2048 * 512 * 2);
  half_t* xprh  = (half_t*)alloc(64 * 1024 * 2);
  half_t* dtwh  = (half_t*)alloc(1024 * 32 * 2);
  half_t* outph = (half_t*)alloc(512 * 1024 * 2);
  // activations
  half_t* xh    = (half_t*)alloc((size_t)M * 512 * 2);
  float*  flipf = (float*) alloc((size_t)M * 512 * 4);
  half_t* fliph = (half_t*)alloc((size_t)M * 512 * 2);
  half_t* xm1h  = (half_t*)alloc((size_t)M * 512 * 2);
  half_t* xp1h  = (half_t*)alloc((size_t)M * 512 * 2);
  float*  h1a   = (float*) alloc((size_t)M * 512 * 4);
  float*  h2a   = (float*) alloc((size_t)M * 512 * 4);
  half_t* h1si  = (half_t*)alloc((size_t)M * 512 * 2);
  half_t* h1sg  = (half_t*)alloc((size_t)M * 512 * 2);
  float*  t1    = (float*) alloc((size_t)M * 512 * 4);
  float*  h1f   = (float*) alloc((size_t)M * 512 * 4);
  half_t* h1sgN = (half_t*)alloc((size_t)M * 512 * 2);
  half_t* s2h   = (half_t*)alloc((size_t)M * 512 * 2);
  float*  h2f   = (float*) alloc((size_t)M * 512 * 4);
  float*  g1f   = (float*) alloc((size_t)M * 512 * 4);
  half_t* g1h   = (half_t*)alloc((size_t)M * 512 * 2);
  half_t* xz16  = (half_t*)alloc((size_t)M * 2048 * 2);   // aliased by gi16
  half_t* xch   = (half_t*)alloc((size_t)M * 1024 * 2);
  float*  projf = (float*) alloc((size_t)M * 64 * 4);
  half_t* projh = (half_t*)alloc((size_t)M * 64 * 2);
  half_t* dt16  = (half_t*)alloc((size_t)M * 1024 * 2);
  half_t* y16   = (half_t*)alloc((size_t)M * 1024 * 2);
  float*  mo    = (float*) alloc((size_t)M * 512 * 4);
  float*  mf    = (float*) alloc((size_t)M * 512 * 4);
  float*  gruo  = (float*) alloc((size_t)M * 512 * 4);
  half_t* comb  = (half_t*)alloc((size_t)M * 512 * 2);
  half_t* gi16  = xz16;  // [M,1536] f16 (mambas finish before GRU)
  (void)ws_size; (void)n_in; (void)in_sizes; (void)out_size;

  auto cast = [&](const float* s, half_t* d, int n) {
    k_cast16<<<(n + 255) / 256, 256, 0, stream>>>(s, d, n);
  };
  cast(W1, w1h, 512 * 512);       cast(W2, w2h, 512 * 512);
  cast(Wp, wph, 512 * 512);       cast(Wsig, wsigh, 512 * 512);
  cast(Wsi, wsih, 512 * 512);     cast(Wih, wihh, 1536 * 512);
  cast(Whh, whhh, 1536 * 512);    cast(inprojW, inph, 2048 * 512);
  cast(xprojW, xprh, 64 * 1024);  cast(dtW, dtwh, 1024 * 32);
  cast(outprojW, outph, 512 * 1024);
  k_wc_split<<<(512 * 512 + 255) / 256, 256, 0, stream>>>(Wc, wc0h, wc1h, wc2h, 512 * 512);

  k_prep<<<(M * 512 + 255) / 256, 256, 0, stream>>>(x, xh, flipf, fliph, xm1h, xp1h, M * 512);

  // h1a = x + x@W1^T + b1 ; h2a = flipped + flipped@W2^T + b2
  gemm(stream, xh, 512, w1h, 512, b1, x, nullptr, h1a, nullptr, 0, nullptr, 0, M, 512, 512);
  gemm(stream, fliph, 512, w2h, 512, b2, flipf, nullptr, h2a, nullptr, 0, nullptr, 0, M, 512, 512);

  // h1 = silu(h1a)@Wsi + bsi + sigmoid(h1a)@Wsig + bsig
  k_act_pair<<<(M * 512 + 255) / 256, 256, 0, stream>>>(h1a, h1si, h1sg, M * 512);
  gemm(stream, h1si, 512, wsih, 512, bsi, nullptr, nullptr, t1, nullptr, 0, nullptr, 0, M, 512, 512);
  gemm(stream, h1sg, 512, wsigh, 512, bsig, t1, nullptr, h1f, h1sgN, 2, nullptr, 0, M, 512, 512);
  // h2 = silu(h2a)@Wsi + bsi + sigmoid(h1_new)@Wsig + bsig
  k_act_pair<<<(M * 512 + 255) / 256, 256, 0, stream>>>(h2a, s2h, nullptr, M * 512);
  gemm(stream, s2h, 512, wsih, 512, bsi, nullptr, nullptr, t1, nullptr, 0, nullptr, 0, M, 512, 512);
  gemm(stream, h1sgN, 512, wsigh, 512, bsig, t1, nullptr, h2f, nullptr, 0, nullptr, 0, M, 512, 512);

  // g1 = conv1d_same(x, Wc, bc) as 3 shifted GEMMs
  gemm(stream, xm1h, 512, wc0h, 512, nullptr, nullptr, nullptr, g1f, nullptr, 0, nullptr, 0, M, 512, 512);
  gemm(stream, xh,   512, wc1h, 512, nullptr, g1f, nullptr, g1f, nullptr, 0, nullptr, 0, M, 512, 512);
  gemm(stream, xp1h, 512, wc2h, 512, bc, g1f, nullptr, g1f, g1h, 0, nullptr, 0, M, 512, 512);

  // mamba(t) twice (output gate fused into out_proj epilogue)
  auto mamba = [&](const half_t* th, const float* gatef, float* mout) {
    gemm(stream, th, 512, inph, 512, nullptr, nullptr, nullptr,
         nullptr, xz16, 0, nullptr, 0, M, 2048, 512);
    k_conv4<<<(M * 1024 + 255) / 256, 256, 0, stream>>>(xz16, convW, convb, xch, M * 1024);
    gemm(stream, xch, 1024, xprh, 1024, nullptr, nullptr, nullptr,
         projf, projh, 0, nullptr, 0, M, 64, 1024);
    gemm(stream, projh, 64, dtwh, 32, dtb, nullptr, nullptr,
         nullptr, dt16, 3, nullptr, 0, M, 1024, 32);
    k_mamba_scan<<<MB_B, 1024, 0, stream>>>(projf, dt16, xch, xz16, Alog, Dp, y16);
    gemm(stream, y16, 1024, outph, 1024, nullptr, nullptr, gatef,
         mout, nullptr, 0, nullptr, 0, M, 512, 1024);
  };
  mamba(xh, h1f, mo);      // mamba_o * (1 + h1)
  mamba(fliph, h2f, mf);   // mamba_f * (1 + h2)

  // GRU: gi = g1@Wih^T, then persistent WMMA recurrence
  gemm(stream, g1h, 512, wihh, 512, nullptr, nullptr, nullptr,
       nullptr, gi16, 0, nullptr, 0, M, 1536, 512);
  k_gru<<<1, 256, 0, stream>>>(whhh, gi16, gruo);

  // combine + final projection
  k_combine<<<(M * 512 + 255) / 256, 256, 0, stream>>>(cw, mo, mf, gruo, comb, M * 512);
  gemm(stream, comb, 512, wph, 512, bp, nullptr, nullptr,
       out, nullptr, 0, nullptr, 0, M, 512, 512);
}
